// GNARLayer_65996467471051
// MI455X (gfx1250) — compile-verified, matrix operand order checked by reading, not run on hardware
//
#include <hip/hip_runtime.h>
#include <hip/hip_bf16.h>

typedef __attribute__((ext_vector_type(16))) __bf16 v16bf;
typedef __attribute__((ext_vector_type(8)))  __bf16 v8bf;
typedef __attribute__((ext_vector_type(2)))  __bf16 v2bf;
typedef __attribute__((ext_vector_type(8)))  float  v8f;
typedef __attribute__((ext_vector_type(4)))  float  v4f;
typedef __attribute__((ext_vector_type(4)))  int    v4i;

#define KDIM 1024
#define TDIM 16384
#define TOUT 16382

#define GLOBAL_AS __attribute__((address_space(1)))
#define LDS_AS    __attribute__((address_space(3)))

// map k (0..31 within a 16x32 bf16 WMMA fragment) -> (half, elem index)
__device__ __forceinline__ void k_map(int k, int& half, int& idx) {
    half = (k >> 3) & 1;                 // k in {8..15,24..31} -> upper 16 lanes
    idx  = (k & 7) + ((k >> 4) << 3);    // +8 for k>=16
}

// 16-byte async global->LDS copy (CDNA5), ASYNCcnt-tracked
__device__ __forceinline__ void async_copy16(void* lds_ptr, const void* gptr) {
#if __has_builtin(__builtin_amdgcn_global_load_async_to_lds_b128)
    __builtin_amdgcn_global_load_async_to_lds_b128(
        (GLOBAL_AS v4i*)gptr, (LDS_AS v4i*)lds_ptr, 0, 0);
#else
    unsigned off = (unsigned)(size_t)lds_ptr;   // low 32 bits of flat shared ptr == LDS offset
    asm volatile("global_load_async_to_lds_b128 %0, %1, off"
                 :: "v"(off), "v"(gptr) : "memory");
#endif
}

__device__ __forceinline__ void wait_async0() {
#if __has_builtin(__builtin_amdgcn_s_wait_asynccnt)
    __builtin_amdgcn_s_wait_asynccnt(0);
#else
    asm volatile("s_wait_asynccnt 0" ::: "memory");
#endif
}

// ---------------------------------------------------------------- K1: A -> bf16
__global__ __launch_bounds__(256) void cvt_bf16_kernel(const float* __restrict__ A,
                                                       __bf16* __restrict__ Ab, int n) {
    int i = blockIdx.x * 256 + threadIdx.x;
    if (i < n) Ab[i] = (__bf16)A[i];
}

// ---------------------------------------------------------------- K2: C = A @ A  (bf16 WMMA, exact for 0/1 data)
__global__ __launch_bounds__(256) void aa_gemm_kernel(const __bf16* __restrict__ Ab,
                                                      float* __restrict__ C) {
    __shared__ __align__(64) __bf16 sA[4096];   // 8 frags * 512
    __shared__ __align__(64) __bf16 sB[4096];

    const int t    = threadIdx.x;
    const int lane = t & 31;
    const int wid  = t >> 5;
    const int wm   = wid >> 1;          // 0..3
    const int wn   = wid & 1;           // 0..1
    const int r0   = blockIdx.y * 128;
    const int n0   = blockIdx.x * 128;

    v8f acc[2][4];
    #pragma unroll
    for (int a = 0; a < 2; ++a)
        #pragma unroll
        for (int b = 0; b < 4; ++b) acc[a][b] = (v8f){};

    for (int kt = 0; kt < 32; ++kt) {
        __syncthreads();
        #pragma unroll
        for (int q = 0; q < 2; ++q) {
            int ch  = t + 256 * q;
            int row = ch >> 2;
            int oct = ch & 3;
            const __bf16* src = Ab + (size_t)(r0 + row) * KDIM + kt * 32 + oct * 8;
            int half = oct & 1;
            int ln   = (row & 15) + (half << 4);
            int frag = row >> 4;
            *(v8bf*)(sA + frag * 512 + ln * 16 + ((oct >> 1) << 3)) = *(const v8bf*)src;
        }
        #pragma unroll
        for (int q = 0; q < 2; ++q) {
            int ch   = t + 256 * q;
            int krow = ch >> 4;
            int cg   = ch & 15;
            int half, idx; k_map(krow, half, idx);
            const __bf16* src = Ab + (size_t)(kt * 32 + krow) * KDIM + n0 + cg * 8;
            #pragma unroll
            for (int e = 0; e < 8; ++e) {
                int cl = cg * 8 + e;
                sB[(cl >> 4) * 512 + ((cl & 15) + (half << 4)) * 16 + idx] = src[e];
            }
        }
        __syncthreads();
        #pragma unroll
        for (int rt = 0; rt < 2; ++rt) {
            v16bf a = *(const v16bf*)(sA + (wm * 2 + rt) * 512 + lane * 16);
            #pragma unroll
            for (int ct = 0; ct < 4; ++ct) {
                v16bf b = *(const v16bf*)(sB + (wn * 4 + ct) * 512 + lane * 16);
                acc[rt][ct] = __builtin_amdgcn_wmma_f32_16x16x32_bf16(
                    false, a, false, b, (short)0, acc[rt][ct], false, false);
            }
        }
    }
    #pragma unroll
    for (int rt = 0; rt < 2; ++rt)
        #pragma unroll
        for (int ct = 0; ct < 4; ++ct) {
            int rowb = r0 + wm * 32 + rt * 16 + ((lane >> 4) << 3);
            int col  = n0 + wn * 64 + ct * 16 + (lane & 15);
            #pragma unroll
            for (int v = 0; v < 8; ++v)
                C[(size_t)(rowb + v) * KDIM + col] = acc[rt][ct][v];
        }
}

// ---------------------------------------------------------------- K3a: per-row inverse degrees
__global__ __launch_bounds__(256) void degrees_kernel(const float* __restrict__ A,
                                                      const float* __restrict__ C,
                                                      float* __restrict__ inv1,
                                                      float* __restrict__ inv2) {
    __shared__ float s1[256], s2[256];
    const int i = blockIdx.x;
    const int t = threadIdx.x;
    float c1 = 0.f, c2 = 0.f;
    for (int j = t; j < KDIM; j += 256) {
        float a  = A[(size_t)i * KDIM + j];
        bool adj = a > 0.5f;
        c1 += adj ? 1.f : 0.f;
        bool two = (C[(size_t)i * KDIM + j] > 0.5f) && !adj && (j != i);
        c2 += two ? 1.f : 0.f;
    }
    s1[t] = c1; s2[t] = c2;
    __syncthreads();
    for (int s = 128; s > 0; s >>= 1) {
        if (t < s) { s1[t] += s1[t + s]; s2[t] += s2[t + s]; }
        __syncthreads();
    }
    if (t == 0) {
        inv1[i] = 1.f / fmaxf(s1[0], 1.f);
        inv2[i] = 1.f / fmaxf(s2[0], 1.f);
    }
}

// ---------------------------------------------------------------- K3b: build stacked [M0;M1] split bf16, pre-swizzled
__global__ __launch_bounds__(256) void build_m_kernel(const float* __restrict__ A,
                                                      const float* __restrict__ C,
                                                      const float* __restrict__ inv1,
                                                      const float* __restrict__ inv2,
                                                      __bf16* __restrict__ Mh,
                                                      __bf16* __restrict__ Ml) {
    int idx = blockIdx.x * 256 + threadIdx.x;  // 0 .. 2*1024*1024-1
    int r = idx >> 10;
    int j = idx & 1023;
    float m;
    if (r < KDIM) {
        m = (A[(size_t)r * KDIM + j] > 0.5f) ? inv1[r] : 0.f;
    } else {
        int i  = r - KDIM;
        float a = A[(size_t)i * KDIM + j];
        m = ((C[(size_t)i * KDIM + j] > 0.5f) && (a <= 0.5f) && (j != i)) ? inv2[i] : 0.f;
    }
    __bf16 h  = (__bf16)m;
    float rem = m - (float)h;
    int half, id; k_map(j & 31, half, id);
    size_t off = ((size_t)((r >> 4) * 32 + (j >> 5)) * 32 + ((r & 15) + (half << 4))) * 16 + id;
    Mh[off] = h;
    Ml[off] = (__bf16)rem;
}

// ---------------------------------------------------------------- Main: fused [M0;M1]@X (split bf16 WMMA) + lag epilogue
// Double-buffered LDS stages; M tiles via async global->LDS; X converted fp32->bf16 hi/lo on the fly.
__global__ __launch_bounds__(256) void gnar_main_kernel(const float* __restrict__ X,
                                                        const float* __restrict__ alpha,
                                                        const float* __restrict__ beta0,
                                                        const float* __restrict__ beta1,
                                                        const __bf16* __restrict__ Mh,
                                                        const __bf16* __restrict__ Ml,
                                                        float* __restrict__ Y) {
    __shared__ __align__(64) unsigned char smem[65536];
    // buffer b (b=0,1) at smem + b*32768: Ah(8KB) Al(8KB) Bh(8KB) Bl(8KB)
    float* sZ = (float*)(smem);                 // 64KB Z tile, reused after K loop

    const int t    = threadIdx.x;
    const int lane = t & 31;
    const int wid  = t >> 5;
    const int wm   = wid >> 1;                  // 0..3
    const int wn   = wid & 1;                   // 0..1
    const int r0   = blockIdx.y * 64;           // node row base
    const int c0   = blockIdx.x * 112;          // output col base (compute 128 cols)
    const int rtb  = r0 >> 4;

    v8f acc[2][4];
    #pragma unroll
    for (int a = 0; a < 2; ++a)
        #pragma unroll
        for (int b = 0; b < 4; ++b) acc[a][b] = (v8f){};

    // ---- per-thread loader geometry
    const int fM   = t >> 5;                                        // M fragment 0..7
    const int rts  = (fM < 4) ? (rtb + fM) : (64 + rtb + (fM - 4)); // stacked rowtile

    float xr[16];                // 2 chunks x (2 rows x 4 cols)

    // issue M async loads for step kt into buffer buf
    auto load_m = [&](int kt, int buf) {
        size_t src = (size_t)(rts * 32 + kt) * 512 + (size_t)lane * 16;
        unsigned char* dAh = smem + buf * 32768 + fM * 1024 + lane * 32;
        async_copy16(dAh,           Mh + src);
        async_copy16(dAh + 16,      Mh + src + 8);
        async_copy16(dAh + 8192,    Ml + src);
        async_copy16(dAh + 8208,    Ml + src + 8);
    };
    // load X fp32 for step kt into registers
    auto load_x = [&](int kt) {
        #pragma unroll
        for (int q = 0; q < 2; ++q) {
            int ch = t + 256 * q;
            int kp = ch >> 5;            // 0..15
            int cg = ch & 31;            // 4-col group
            int gk = kt * 32 + 2 * kp;
            int cbase = c0 + cg * 4;
            const float* xp0 = X + (size_t)gk * TDIM + cbase;
            const float* xp1 = xp0 + TDIM;
            if (cbase + 3 < TDIM) {
                v4f a = *(const v4f*)xp0;
                v4f b = *(const v4f*)xp1;
                #pragma unroll
                for (int e = 0; e < 4; ++e) { xr[q*8 + e] = a[e]; xr[q*8 + 4 + e] = b[e]; }
            } else {
                #pragma unroll
                for (int e = 0; e < 4; ++e) {
                    bool ok = (cbase + e) < TDIM;
                    xr[q*8 + e]     = ok ? xp0[e] : 0.f;
                    xr[q*8 + 4 + e] = ok ? xp1[e] : 0.f;
                }
            }
        }
    };
    // split + store X registers into buffer buf (packed 2xbf16 b32 stores)
    auto store_x = [&](int buf) {
        __bf16* bh = (__bf16*)(smem + buf * 32768 + 16384);
        __bf16* bl = (__bf16*)(smem + buf * 32768 + 24576);
        #pragma unroll
        for (int q = 0; q < 2; ++q) {
            int ch = t + 256 * q;
            int kp = ch >> 5;
            int cg = ch & 31;
            int half = (kp >> 2) & 1;
            int idx  = 2 * (kp & 3) + ((kp >> 3) << 3);   // even; rows 2kp,2kp+1 -> idx,idx+1
            #pragma unroll
            for (int e = 0; e < 4; ++e) {
                int cl  = cg * 4 + e;
                int off = (cl >> 4) * 512 + ((cl & 15) + (half << 4)) * 16 + idx;
                float x0 = xr[q*8 + e], x1 = xr[q*8 + 4 + e];
                __bf16 h0 = (__bf16)x0, h1 = (__bf16)x1;
                *(v2bf*)(bh + off) = (v2bf){h0, h1};
                *(v2bf*)(bl + off) = (v2bf){(__bf16)(x0 - (float)h0), (__bf16)(x1 - (float)h1)};
            }
        }
    };

    // ---- prologue: fill buffer 0
    load_m(0, 0);
    load_x(0);
    store_x(0);
    wait_async0();
    __syncthreads();

    for (int kt = 0; kt < 32; ++kt) {
        const int cur = kt & 1;
        const bool more = (kt + 1) < 32;
        if (more) { load_m(kt + 1, cur ^ 1); load_x(kt + 1); }

        const __bf16* sAh = (const __bf16*)(smem + cur * 32768);
        const __bf16* sAl = sAh + 4096;
        const __bf16* sBh = sAh + 8192;
        const __bf16* sBl = sAh + 12288;
        #pragma unroll
        for (int rt = 0; rt < 2; ++rt) {
            v16bf ah = *(const v16bf*)(sAh + (wm * 2 + rt) * 512 + lane * 16);
            v16bf al = *(const v16bf*)(sAl + (wm * 2 + rt) * 512 + lane * 16);
            #pragma unroll
            for (int ct = 0; ct < 4; ++ct) {
                v16bf bh = *(const v16bf*)(sBh + (wn * 4 + ct) * 512 + lane * 16);
                v16bf bl = *(const v16bf*)(sBl + (wn * 4 + ct) * 512 + lane * 16);
                acc[rt][ct] = __builtin_amdgcn_wmma_f32_16x16x32_bf16(
                    false, ah, false, bh, (short)0, acc[rt][ct], false, false);
                acc[rt][ct] = __builtin_amdgcn_wmma_f32_16x16x32_bf16(
                    false, ah, false, bl, (short)0, acc[rt][ct], false, false);
                acc[rt][ct] = __builtin_amdgcn_wmma_f32_16x16x32_bf16(
                    false, al, false, bh, (short)0, acc[rt][ct], false, false);
            }
        }
        if (more) { store_x(cur ^ 1); wait_async0(); }
        __syncthreads();
    }

    // --- spill Z (128 stacked rows x 128 cols) to LDS
    #pragma unroll
    for (int rt = 0; rt < 2; ++rt)
        #pragma unroll
        for (int ct = 0; ct < 4; ++ct) {
            int rowb = wm * 32 + rt * 16 + ((lane >> 4) << 3);
            int col  = wn * 64 + ct * 16 + (lane & 15);
            #pragma unroll
            for (int v = 0; v < 8; ++v)
                sZ[(rowb + v) * 128 + col] = acc[rt][ct][v];
        }
    __syncthreads();

    // --- lag epilogue
    const float a0 = alpha[0], a1 = alpha[1];
    const float b00 = beta0[0], b01 = beta0[1];
    const float b10 = beta1[0], b11 = beta1[1];
    const int tx = t & 127;
    const int ty = t >> 7;
    if (tx < 112) {
        int cg = c0 + tx;
        if (cg < TOUT) {
            for (int r = ty; r < 64; r += 2) {
                int gi = r0 + r;
                float z0a = sZ[r * 128 + tx];
                float z0b = sZ[r * 128 + tx + 1];
                float z1a = sZ[(64 + r) * 128 + tx];
                float z1b = sZ[(64 + r) * 128 + tx + 1];
                float xa  = X[(size_t)gi * TDIM + cg];
                float xb  = X[(size_t)gi * TDIM + cg + 1];
                Y[(size_t)gi * TOUT + cg] =
                    a0 * xb + a1 * xa + b00 * z0b + b01 * z0a + b10 * z1b + b11 * z1a;
            }
        }
    }
}

// ---------------------------------------------------------------- launcher
extern "C" void kernel_launch(void* const* d_in, const int* in_sizes, int n_in,
                              void* d_out, int out_size, void* d_ws, size_t ws_size,
                              hipStream_t stream) {
    const float* X     = (const float*)d_in[0];
    const float* A     = (const float*)d_in[1];
    const float* alpha = (const float*)d_in[2];
    const float* beta0 = (const float*)d_in[3];
    const float* beta1 = (const float*)d_in[4];
    float* Y           = (float*)d_out;

    unsigned char* ws = (unsigned char*)d_ws;
    __bf16* Abf  = (__bf16*)(ws);                       // 2 MB
    float*  C    = (float*) (ws + 0x200000);            // 4 MB
    float*  inv1 = (float*) (ws + 0x600000);            // 4 KB
    float*  inv2 = (float*) (ws + 0x601000);            // 4 KB
    __bf16* Mh   = (__bf16*)(ws + 0x610000);            // 4 MB (swizzled)
    __bf16* Ml   = (__bf16*)(ws + 0xA10000);            // 4 MB (swizzled)

    cvt_bf16_kernel<<<4096, 256, 0, stream>>>(A, Abf, KDIM * KDIM);
    aa_gemm_kernel<<<dim3(8, 8), 256, 0, stream>>>(Abf, C);
    degrees_kernel<<<KDIM, 256, 0, stream>>>(A, C, inv1, inv2);
    build_m_kernel<<<8192, 256, 0, stream>>>(A, C, inv1, inv2, Mh, Ml);
    gnar_main_kernel<<<dim3(147, 16), 256, 0, stream>>>(X, alpha, beta0, beta1, Mh, Ml, Y);
}
